// IBertSelfAttention_18880676233622
// MI455X (gfx1250) — 8-bit matrix operand order repaired and checked by reading, compiled, not run on hardware
//
#include <hip/hip_runtime.h>
#include <hip/hip_bf16.h>
#include <math.h>
#include <stdint.h>

#define S_LEN 2048
#define HID   768
#define NHEAD 12
#define HDIM  64

typedef __attribute__((ext_vector_type(8))) int v8i;

enum {
  P_MINQ = 0, P_MAXQ, P_MINK, P_MAXK, P_MINV, P_MAXV, P_MINE, P_MAXE,
  P_MINC, P_MAXC,
  P_SQ, P_SK, P_SV,
  P_SSC,
  P_X0, P_BSM, P_CSM,
  P_ES,
  P_S16, P_EOS,
  P_CTXPRE, P_CTXS
};

__device__ __forceinline__ v8i v8zero() {
  v8i c;
#pragma unroll
  for (int i = 0; i < 8; ++i) c[i] = 0;
  return c;
}
__device__ __forceinline__ float clampf(float x, float lo, float hi) {
  return fminf(fmaxf(x, lo), hi);
}
__device__ __forceinline__ int imax(int a, int b) { return a > b ? a : b; }
__device__ __forceinline__ int imin(int a, int b) { return a < b ? a : b; }

__device__ __forceinline__ unsigned fflip(float f) {
  unsigned u = __float_as_uint(f);
  return (u & 0x80000000u) ? ~u : (u | 0x80000000u);
}
__device__ __forceinline__ float funflip(unsigned u) {
  u = (u & 0x80000000u) ? (u ^ 0x80000000u) : ~u;
  return __uint_as_float(u);
}

__device__ __forceinline__ v8i load_a_frag(const char* base, int ld) {
  int lane = threadIdx.x & 31;
  int m = lane & 15, half = lane >> 4;
  const char* rp = base + (size_t)m * ld;
  v8i a;
#pragma unroll
  for (int v = 0; v < 8; ++v) {
    int k = ((v >> 1) << 4) + ((v & 1) << 2) + (half << 3);
    a[v] = *(const int*)(rp + k);
  }
  return a;
}
__device__ __forceinline__ v8i load_b_frag(const char* base, int ld) {
  int lane = threadIdx.x & 31;
  int n = lane & 15, kh = lane >> 4;
  const char* cp = base + (size_t)n * ld;
  v8i b;
#pragma unroll
  for (int v = 0; v < 8; ++v) {
    int k = ((v >> 1) << 4) + ((v & 1) << 2) + (kh << 3);
    b[v] = *(const int*)(cp + k);
  }
  return b;
}

__device__ __forceinline__ float int_exp(float acc, float rmax, float x0f, float bsm, float csm) {
  float xi = acc - rmax;
  xi = fmaxf(xi, 30.0f * x0f);
  float qf = floorf(xi / x0f);
  float r  = xi - x0f * qf;
  float zz = (r + bsm) * r + csm;
  return fmaxf(floorf(zz * exp2f(30.0f - qf)), 0.0f);
}

__global__ void k_init(unsigned* pu) {
  if (threadIdx.x == 0 && blockIdx.x == 0) {
    pu[P_MINQ] = 0xFFFFFFFFu; pu[P_MAXQ] = 0u;
    pu[P_MINK] = 0xFFFFFFFFu; pu[P_MAXK] = 0u;
    pu[P_MINV] = 0xFFFFFFFFu; pu[P_MAXV] = 0u;
    pu[P_MINE] = 0xFFFFFFFFu; pu[P_MAXE] = 0u;
    ((int*)pu)[P_MINC] = 0x7FFFFFFF;
    ((int*)pu)[P_MAXC] = (int)0x80000000;
  }
}

__global__ void k_quant_w(const float* __restrict__ W, const float* __restrict__ bias,
                          const float* __restrict__ s_in,
                          char* __restrict__ w8, float* __restrict__ bs, int* __restrict__ bi) {
  __shared__ float smn[128], smx[128];
  __shared__ float sscale;
  int row = blockIdx.x, t = threadIdx.x;
  const float* wr = W + (size_t)row * HID;
  float mn = 1e30f, mx = -1e30f;
  for (int j = t; j < HID; j += 128) { float v = wr[j]; mn = fminf(mn, v); mx = fmaxf(mx, v); }
  smn[t] = mn; smx[t] = mx; __syncthreads();
  for (int s = 64; s > 0; s >>= 1) {
    if (t < s) { smn[t] = fminf(smn[t], smn[t + s]); smx[t] = fmaxf(smx[t], smx[t + s]); }
    __syncthreads();
  }
  if (t == 0) {
    float m  = fmaxf(fmaxf(fabsf(smn[0]), fabsf(smx[0])), 1e-8f);
    float sc = m / 127.0f;
    sscale = sc;
    float bsf = sc * s_in[0];
    bs[row] = bsf;
    float bq = clampf(rintf(bias[row] / bsf), -2147483648.0f, 2147483520.0f);
    bi[row] = (int)bq;
  }
  __syncthreads();
  float sc = sscale;
  for (int j = t; j < HID; j += 128) {
    float q = clampf(rintf(wr[j] / sc), -128.0f, 127.0f);
    w8[(size_t)row * HID + j] = (char)(int)q;
  }
}

__global__ void k_quant_x(const float* __restrict__ x, const float* __restrict__ s_in,
                          char* __restrict__ x8, int n) {
  int i = blockIdx.x * blockDim.x + threadIdx.x;
  if (i >= n) return;
  float q = clampf(rintf(x[i] / s_in[0]), -128.0f, 127.0f);
  x8[i] = (char)(int)q;
}

__global__ void __launch_bounds__(256) k_gemm(
    const char* __restrict__ x8, const char* __restrict__ w8,
    const int* __restrict__ bi, const float* __restrict__ bs,
    int* __restrict__ z, unsigned* __restrict__ pmin, unsigned* __restrict__ pmax) {
  int wid = blockIdx.x * (blockDim.x >> 5) + (threadIdx.x >> 5);
  const int MT = S_LEN / 32, NT = HID / 32;
  if (wid >= MT * NT) return;
  int m0 = (wid % MT) * 32, n0 = (wid / MT) * 32;
  v8i acc[2][2];
  acc[0][0] = v8zero(); acc[0][1] = v8zero(); acc[1][0] = v8zero(); acc[1][1] = v8zero();
  for (int kc = 0; kc < HID; kc += 64) {
    __builtin_prefetch(x8 + (size_t)m0 * HID + kc + 64, 0, 0);
    v8i a0 = load_a_frag(x8 + (size_t)m0 * HID + kc, HID);
    v8i a1 = load_a_frag(x8 + (size_t)(m0 + 16) * HID + kc, HID);
    v8i b0 = load_b_frag(w8 + (size_t)n0 * HID + kc, HID);
    v8i b1 = load_b_frag(w8 + (size_t)(n0 + 16) * HID + kc, HID);
    acc[0][0] = __builtin_amdgcn_wmma_i32_16x16x64_iu8(true, a0, true, b0, acc[0][0], false, false);
    acc[0][1] = __builtin_amdgcn_wmma_i32_16x16x64_iu8(true, a0, true, b1, acc[0][1], false, false);
    acc[1][0] = __builtin_amdgcn_wmma_i32_16x16x64_iu8(true, a1, true, b0, acc[1][0], false, false);
    acc[1][1] = __builtin_amdgcn_wmma_i32_16x16x64_iu8(true, a1, true, b1, acc[1][1], false, false);
  }
  int lane = threadIdx.x & 31;
  int n = lane & 15, half = lane >> 4;
  float lmn = 1e30f, lmx = -1e30f;
#pragma unroll
  for (int tm = 0; tm < 2; ++tm)
#pragma unroll
    for (int tn = 0; tn < 2; ++tn)
#pragma unroll
      for (int r = 0; r < 8; ++r) {
        int row = m0 + tm * 16 + r + 8 * half;
        int col = n0 + tn * 16 + n;
        int zv  = acc[tm][tn][r] + bi[col];
        z[(size_t)row * HID + col] = zv;
        float f = (float)zv * bs[col];
        lmn = fminf(lmn, f); lmx = fmaxf(lmx, f);
      }
#pragma unroll
  for (int off = 16; off > 0; off >>= 1) {
    lmn = fminf(lmn, __shfl_xor(lmn, off, 32));
    lmx = fmaxf(lmx, __shfl_xor(lmx, off, 32));
  }
  if (lane == 0) { atomicMin(pmin, fflip(lmn)); atomicMax(pmax, fflip(lmx)); }
}

__global__ void k_params1(unsigned* pu) {
  if (threadIdx.x || blockIdx.x) return;
  float* pf = (float*)pu;
  float sq = fmaxf(fmaxf(fabsf(funflip(pu[P_MINQ])), fabsf(funflip(pu[P_MAXQ]))), 1e-8f) / 127.0f;
  float sk = fmaxf(fmaxf(fabsf(funflip(pu[P_MINK])), fabsf(funflip(pu[P_MAXK]))), 1e-8f) / 127.0f;
  float sv = fmaxf(fmaxf(fabsf(funflip(pu[P_MINV])), fabsf(funflip(pu[P_MAXV]))), 1e-8f) / 127.0f;
  pf[P_SQ] = sq; pf[P_SK] = sk; pf[P_SV] = sv;
  float ssc = sq * sk / 8.0f;
  pf[P_SSC] = ssc;
  pf[P_X0]  = floorf(-0.6931f / ssc);
  pf[P_BSM] = floorf((float)(0.96963238 / 0.35815147) / ssc);
  pf[P_CSM] = floorf((float)(1.0 / 0.35815147) / (ssc * ssc));
  pf[P_ES]  = 0.35815147f * ssc * ssc * 0x1p-30f;
}

__global__ void k_requant(const int* __restrict__ z, const float* __restrict__ bs,
                          const float* __restrict__ pf, int slot,
                          char* __restrict__ out8, int transpose_v, int n) {
  int i = blockIdx.x * blockDim.x + threadIdx.x;
  if (i >= n) return;
  float sact = pf[slot];
  int s = i / HID, col = i % HID;
  float q = clampf(rintf((float)z[i] * bs[col] / sact), -128.0f, 127.0f);
  char qc = (char)(int)q;
  if (transpose_v) out8[(size_t)col * S_LEN + s] = qc;
  else             out8[i] = qc;
}

__global__ void __launch_bounds__(256) k_rowmax(const char* __restrict__ q8,
                                                const char* __restrict__ k8,
                                                float* __restrict__ rowmax) {
  int wid = blockIdx.x * (blockDim.x >> 5) + (threadIdx.x >> 5);
  if (wid >= NHEAD * (S_LEN / 16)) return;
  int h = wid / (S_LEN / 16);
  int s0 = (wid % (S_LEN / 16)) * 16;
  v8i a = load_a_frag(q8 + (size_t)s0 * HID + h * HDIM, HID);
  int mx[8];
#pragma unroll
  for (int r = 0; r < 8; ++r) mx[r] = (int)0x80000000;
  for (int tt = 0; tt < S_LEN / 16; ++tt) {
    v8i b = load_b_frag(k8 + (size_t)(tt * 16) * HID + h * HDIM, HID);
    v8i c = __builtin_amdgcn_wmma_i32_16x16x64_iu8(true, a, true, b, v8zero(), false, false);
#pragma unroll
    for (int r = 0; r < 8; ++r) mx[r] = imax(mx[r], c[r]);
  }
  int lane = threadIdx.x & 31, half = lane >> 4;
#pragma unroll
  for (int r = 0; r < 8; ++r)
#pragma unroll
    for (int off = 1; off < 16; off <<= 1) mx[r] = imax(mx[r], __shfl_xor(mx[r], off, 32));
  if ((lane & 15) == 0)
#pragma unroll
    for (int r = 0; r < 8; ++r) rowmax[h * S_LEN + s0 + r + 8 * half] = (float)mx[r];
}

__global__ void __launch_bounds__(256) k_expminmax(const char* __restrict__ q8,
                                                   const char* __restrict__ k8,
                                                   const float* __restrict__ rowmax,
                                                   const float* __restrict__ pf,
                                                   unsigned* __restrict__ pmin,
                                                   unsigned* __restrict__ pmax) {
  int wid = blockIdx.x * (blockDim.x >> 5) + (threadIdx.x >> 5);
  if (wid >= NHEAD * (S_LEN / 16)) return;
  int h = wid / (S_LEN / 16);
  int s0 = (wid % (S_LEN / 16)) * 16;
  int lane = threadIdx.x & 31, half = lane >> 4;
  float x0f = pf[P_X0], bsm = pf[P_BSM], csm = pf[P_CSM];
  v8i a = load_a_frag(q8 + (size_t)s0 * HID + h * HDIM, HID);
  float rmx[8];
#pragma unroll
  for (int r = 0; r < 8; ++r) rmx[r] = rowmax[h * S_LEN + s0 + r + 8 * half];
  float lmn = 1e38f, lmx = -1e38f;
  for (int tt = 0; tt < S_LEN / 16; ++tt) {
    v8i b = load_b_frag(k8 + (size_t)(tt * 16) * HID + h * HDIM, HID);
    v8i c = __builtin_amdgcn_wmma_i32_16x16x64_iu8(true, a, true, b, v8zero(), false, false);
#pragma unroll
    for (int r = 0; r < 8; ++r) {
      float e = int_exp((float)c[r], rmx[r], x0f, bsm, csm);
      lmn = fminf(lmn, e); lmx = fmaxf(lmx, e);
    }
  }
#pragma unroll
  for (int off = 16; off > 0; off >>= 1) {
    lmn = fminf(lmn, __shfl_xor(lmn, off, 32));
    lmx = fmaxf(lmx, __shfl_xor(lmx, off, 32));
  }
  if (lane == 0) { atomicMin(pmin, fflip(lmn)); atomicMax(pmax, fflip(lmx)); }
}

__global__ void k_params2(unsigned* pu) {
  if (threadIdx.x || blockIdx.x) return;
  float* pf = (float*)pu;
  float es  = pf[P_ES];
  float mn  = funflip(pu[P_MINE]) * es;
  float mx  = funflip(pu[P_MAXE]) * es;
  float s16 = fmaxf(fmaxf(fabsf(mn), fabsf(mx)), 1e-8f) / 32767.0f;
  pf[P_S16] = s16;
  pf[P_EOS] = es / s16;
}

__global__ void __launch_bounds__(256) k_rowsum(const char* __restrict__ q8,
                                                const char* __restrict__ k8,
                                                const float* __restrict__ rowmax,
                                                const float* __restrict__ pf,
                                                float* __restrict__ factor) {
  int wid = blockIdx.x * (blockDim.x >> 5) + (threadIdx.x >> 5);
  if (wid >= NHEAD * (S_LEN / 16)) return;
  int h = wid / (S_LEN / 16);
  int s0 = (wid % (S_LEN / 16)) * 16;
  int lane = threadIdx.x & 31, half = lane >> 4;
  float x0f = pf[P_X0], bsm = pf[P_BSM], csm = pf[P_CSM], eos = pf[P_EOS];
  v8i a = load_a_frag(q8 + (size_t)s0 * HID + h * HDIM, HID);
  float rmx[8], sum[8];
#pragma unroll
  for (int r = 0; r < 8; ++r) { rmx[r] = rowmax[h * S_LEN + s0 + r + 8 * half]; sum[r] = 0.0f; }
  for (int tt = 0; tt < S_LEN / 16; ++tt) {
    v8i b = load_b_frag(k8 + (size_t)(tt * 16) * HID + h * HDIM, HID);
    v8i c = __builtin_amdgcn_wmma_i32_16x16x64_iu8(true, a, true, b, v8zero(), false, false);
#pragma unroll
    for (int r = 0; r < 8; ++r) {
      float e   = int_exp((float)c[r], rmx[r], x0f, bsm, csm);
      float e16 = clampf(rintf(e * eos), -32768.0f, 32767.0f);
      sum[r] += e16;
    }
  }
#pragma unroll
  for (int r = 0; r < 8; ++r)
#pragma unroll
    for (int off = 1; off < 16; off <<= 1) sum[r] += __shfl_xor(sum[r], off, 32);
  if ((lane & 15) == 0)
#pragma unroll
    for (int r = 0; r < 8; ++r)
      factor[h * S_LEN + s0 + r + 8 * half] = floorf(4294967296.0f / fmaxf(sum[r], 1.0f));
}

__global__ void __launch_bounds__(256) k_ctx(const char* __restrict__ q8,
                                             const char* __restrict__ k8,
                                             const char* __restrict__ v8t,
                                             const float* __restrict__ rowmax,
                                             const float* __restrict__ factor,
                                             const float* __restrict__ pf,
                                             int* __restrict__ zc,
                                             int* __restrict__ pmin, int* __restrict__ pmax) {
  __shared__ __align__(16) unsigned char smem[8 * 16 * 64];
  int wslot = threadIdx.x >> 5;
  unsigned char* sm = smem + wslot * (16 * 64);
  int wid = blockIdx.x * (blockDim.x >> 5) + wslot;
  if (wid >= NHEAD * (S_LEN / 16)) return;
  int h = wid / (S_LEN / 16);
  int s0 = (wid % (S_LEN / 16)) * 16;
  int lane = threadIdx.x & 31, n = lane & 15, half = lane >> 4;
  float x0f = pf[P_X0], bsm = pf[P_BSM], csm = pf[P_CSM], eos = pf[P_EOS];
  v8i a = load_a_frag(q8 + (size_t)s0 * HID + h * HDIM, HID);
  float rmx[8], fac[8];
#pragma unroll
  for (int r = 0; r < 8; ++r) {
    rmx[r] = rowmax[h * S_LEN + s0 + r + 8 * half];
    fac[r] = factor[h * S_LEN + s0 + r + 8 * half];
  }
  v8i cc[4];
#pragma unroll
  for (int db = 0; db < 4; ++db) cc[db] = v8zero();

  for (int tc = 0; tc < S_LEN / 64; ++tc) {
    asm volatile("s_wait_dscnt 0" ::: "memory");
#pragma unroll
    for (int st = 0; st < 4; ++st) {
      int t0 = tc * 64 + st * 16;
      v8i b = load_b_frag(k8 + (size_t)t0 * HID + h * HDIM, HID);
      v8i c = __builtin_amdgcn_wmma_i32_16x16x64_iu8(true, a, true, b, v8zero(), false, false);
#pragma unroll
      for (int r = 0; r < 8; ++r) {
        float e   = int_exp((float)c[r], rmx[r], x0f, bsm, csm);
        float e16 = clampf(rintf(e * eos), -32768.0f, 32767.0f);
        float p   = floorf(e16 * fac[r] * 0x1p-24f);
        p = clampf(p, 0.0f, 255.0f);
        sm[(r + 8 * half) * 64 + st * 16 + n] = (unsigned char)(int)p;
      }
    }
    asm volatile("s_wait_dscnt 0" ::: "memory");
    v8i pa = load_a_frag((const char*)sm, 64);
#pragma unroll
    for (int db = 0; db < 4; ++db) {
      v8i b = load_b_frag(v8t + ((size_t)(h * HDIM + db * 16)) * S_LEN + tc * 64, S_LEN);
      cc[db] = __builtin_amdgcn_wmma_i32_16x16x64_iu8(false, pa, true, b, cc[db], false, false);
    }
  }

  int lmn = 0x7FFFFFFF, lmx = (int)0x80000000;
#pragma unroll
  for (int db = 0; db < 4; ++db)
#pragma unroll
    for (int r = 0; r < 8; ++r) {
      int row = s0 + r + 8 * half;
      int col = h * HDIM + db * 16 + n;
      int v = cc[db][r];
      zc[(size_t)row * HID + col] = v;
      lmn = imin(lmn, v); lmx = imax(lmx, v);
    }
#pragma unroll
  for (int off = 16; off > 0; off >>= 1) {
    lmn = imin(lmn, __shfl_xor(lmn, off, 32));
    lmx = imax(lmx, __shfl_xor(lmx, off, 32));
  }
  if (lane == 0) { atomicMin(pmin, lmn); atomicMax(pmax, lmx); }
}

__global__ void k_params3(unsigned* pu, float* scale_out) {
  if (threadIdx.x || blockIdx.x) return;
  float* pf = (float*)pu;
  int* pi = (int*)pu;
  float pre = pf[P_SV] / 256.0f;
  float mn = (float)pi[P_MINC] * pre;
  float mx = (float)pi[P_MAXC] * pre;
  float cs = fmaxf(fmaxf(fabsf(mn), fabsf(mx)), 1e-8f) / 127.0f;
  pf[P_CTXPRE] = pre; pf[P_CTXS] = cs;
  *scale_out = cs;
}

__global__ void k_out(const int* __restrict__ zc, const float* __restrict__ pf,
                      float* __restrict__ out, int n) {
  int i = blockIdx.x * blockDim.x + threadIdx.x;
  if (i >= n) return;
  float pre = pf[P_CTXPRE], cs = pf[P_CTXS];
  float q = clampf(rintf((float)zc[i] * pre / cs), -128.0f, 127.0f);
  out[i] = q * cs;
}

extern "C" void kernel_launch(void* const* d_in, const int* in_sizes, int n_in,
                              void* d_out, int out_size, void* d_ws, size_t ws_size,
                              hipStream_t stream) {
  (void)in_sizes; (void)n_in; (void)out_size; (void)ws_size;
  const float* x   = (const float*)d_in[0];
  const float* sin = (const float*)d_in[1];
  const float* Wq  = (const float*)d_in[2];
  const float* bq  = (const float*)d_in[3];
  const float* Wk  = (const float*)d_in[4];
  const float* bk  = (const float*)d_in[5];
  const float* Wv  = (const float*)d_in[6];
  const float* bv  = (const float*)d_in[7];
  float* out = (float*)d_out;

  char* ws = (char*)d_ws;
  size_t off = 0;
  auto take = [&](size_t bytes) { size_t o = off; off += (bytes + 255) & ~(size_t)255; return o; };

  const size_t NXH = (size_t)S_LEN * HID;
  unsigned* prm = (unsigned*)(ws + take(256));
  char* w8q = ws + take((size_t)HID * HID);
  char* w8k = ws + take((size_t)HID * HID);
  char* w8v = ws + take((size_t)HID * HID);
  float* bsq = (float*)(ws + take(HID * 4));
  float* bsk = (float*)(ws + take(HID * 4));
  float* bsv = (float*)(ws + take(HID * 4));
  int* biq = (int*)(ws + take(HID * 4));
  int* bik = (int*)(ws + take(HID * 4));
  int* biv = (int*)(ws + take(HID * 4));
  char* x8 = ws + take(NXH);
  int* zq = (int*)(ws + take(NXH * 4));
  int* zk = (int*)(ws + take(NXH * 4));
  int* zv = (int*)(ws + take(NXH * 4));
  char* q8  = ws + take(NXH);
  char* k8  = ws + take(NXH);
  char* v8t = ws + take(NXH);
  float* rmax = (float*)(ws + take((size_t)NHEAD * S_LEN * 4));
  float* fact = (float*)(ws + take((size_t)NHEAD * S_LEN * 4));
  int* zc = (int*)(ws + take(NXH * 4));

  const float* pf = (const float*)prm;
  int nXH = (int)NXH;

  k_init<<<1, 32, 0, stream>>>(prm);
  k_quant_w<<<HID, 128, 0, stream>>>(Wq, bq, sin, w8q, bsq, biq);
  k_quant_w<<<HID, 128, 0, stream>>>(Wk, bk, sin, w8k, bsk, bik);
  k_quant_w<<<HID, 128, 0, stream>>>(Wv, bv, sin, w8v, bsv, biv);
  k_quant_x<<<nXH / 256, 256, 0, stream>>>(x, sin, x8, nXH);

  int gemmBlocks = (S_LEN / 32) * (HID / 32) / 8;
  k_gemm<<<gemmBlocks, 256, 0, stream>>>(x8, w8q, biq, bsq, zq, prm + P_MINQ, prm + P_MAXQ);
  k_gemm<<<gemmBlocks, 256, 0, stream>>>(x8, w8k, bik, bsk, zk, prm + P_MINK, prm + P_MAXK);
  k_gemm<<<gemmBlocks, 256, 0, stream>>>(x8, w8v, biv, bsv, zv, prm + P_MINV, prm + P_MAXV);
  k_params1<<<1, 1, 0, stream>>>(prm);
  k_requant<<<nXH / 256, 256, 0, stream>>>(zq, bsq, pf, P_SQ, q8, 0, nXH);
  k_requant<<<nXH / 256, 256, 0, stream>>>(zk, bsk, pf, P_SK, k8, 0, nXH);
  k_requant<<<nXH / 256, 256, 0, stream>>>(zv, bsv, pf, P_SV, v8t, 1, nXH);

  int attnBlocks = NHEAD * (S_LEN / 16) / 8;
  k_rowmax<<<attnBlocks, 256, 0, stream>>>(q8, k8, rmax);
  k_expminmax<<<attnBlocks, 256, 0, stream>>>(q8, k8, rmax, pf, prm + P_MINE, prm + P_MAXE);
  k_params2<<<1, 1, 0, stream>>>(prm);
  k_rowsum<<<attnBlocks, 256, 0, stream>>>(q8, k8, rmax, pf, fact);
  k_ctx<<<attnBlocks, 256, 0, stream>>>(q8, k8, v8t, rmax, fact, pf, zc,
                                        (int*)prm + P_MINC, (int*)prm + P_MAXC);
  k_params3<<<1, 1, 0, stream>>>(prm, out + NXH);
  k_out<<<nXH / 256, 256, 0, stream>>>(zc, pf, out, nXH);
}
